// Generator_39900246180055
// MI455X (gfx1250) — matrix operand order repaired and checked by reading, not compiled, not run on hardware
//
#include <hip/hip_runtime.h>
#include <hip/hip_bf16.h>

#define RB 8
#define SEQ 1024
#define NOISE 128
#define TF 6
#define EMB 320
#define NHEADS 5
#define HD 64
#define GEN 2
#define FF 1280
#define ROWS (RB*SEQ)
#define QKD 192   // qx|qh|qt concatenated head dim (3*64)

typedef _Float16 h16;
typedef _Float16 h16v16 __attribute__((ext_vector_type(16)));
typedef _Float16 h16v8  __attribute__((ext_vector_type(8)));
typedef float    f32v8  __attribute__((ext_vector_type(8)));
typedef unsigned int u32v4 __attribute__((ext_vector_type(4)));
typedef int i32v8 __attribute__((ext_vector_type(8)));
typedef int i32v4 __attribute__((ext_vector_type(4)));

typedef __attribute__((address_space(1))) i32v4 glb_v4i;
typedef __attribute__((address_space(3))) i32v4 lds_v4i;

// ---- CDNA5 feature probes (guarded so the file always compiles) --------------------
#if __has_builtin(__builtin_amdgcn_global_load_async_to_lds_b128) && \
    __has_builtin(__builtin_amdgcn_s_wait_asynccnt)
#define HAVE_ASYNC 1
#else
#define HAVE_ASYNC 0
#endif
#if HAVE_ASYNC && __has_builtin(__builtin_amdgcn_tensor_load_to_lds) && \
    __has_builtin(__builtin_amdgcn_s_wait_tensorcnt)
#define HAVE_TDM 1
#else
#define HAVE_TDM 0
#endif

#if HAVE_ASYNC
__device__ __forceinline__ void async_b128(const void* g, void* l) {
  __builtin_amdgcn_global_load_async_to_lds_b128(
      (glb_v4i*)(uintptr_t)g,            // AS(1) v4i*, value-preserving int cast
      (lds_v4i*)l,                       // generic->AS(3) addrspacecast
      0, 0);
}
#endif

#if HAVE_TDM
// One-shot 2D TDM tile load: rows x row_halves (f16), source leading dim ld_halves.
// D# layout per CDNA5 ISA (2D: groups 2/3 zero).
// clang-23 6-arg form: (u32x4 g0, i32x8 g1, i32x4, i32x4, i32x8, i32 cpol)
__device__ __forceinline__ void tdm_load_2d(const h16* gsrc, h16* lds_dst,
                                            int rows, int row_halves, int ld_halves) {
  unsigned long long ga = (unsigned long long)(uintptr_t)gsrc;
  unsigned int lo =
      (unsigned int)(uintptr_t)(__attribute__((address_space(3))) h16*)lds_dst;
  u32v4 g0;
  g0[0] = 0x1u;                                           // count=1 (user descriptor)
  g0[1] = lo;                                             // lds_addr (bytes)
  g0[2] = (unsigned int)ga;                               // global_addr[31:0]
  g0[3] = ((unsigned int)(ga >> 32) & 0x01FFFFFFu) | 0x80000000u; // addr[56:32]|type=2
  i32v8 g1;
  g1[0] = 0x10000;                        // workgroup_mask=0 (no cluster), data_size=2B
  g1[1] = (int)0xFFFF0000u;               // tensor_dim0 = huge (no OOB clipping)
  g1[2] = (int)0xFFFFFFFFu;               // tensor_dim0 hi | tensor_dim1 lo
  g1[3] = (int)(0xFFFFu | ((unsigned int)row_halves << 16)); // tensor_dim1 hi|tile_dim0
  g1[4] = rows;                           // tile_dim1 | tile_dim2=0
  g1[5] = ld_halves;                      // tensor_dim0_stride[31:0] (data_size units)
  g1[6] = 0;
  g1[7] = 0;
  i32v4 gz4 = {0, 0, 0, 0};
  i32v8 gz8 = {0, 0, 0, 0, 0, 0, 0, 0};
  __builtin_amdgcn_tensor_load_to_lds(g0, g1, gz4, gz4, gz8, 0);
}
#endif

// ---------------- WMMA fragment helpers (layouts per CDNA5 ISA §7.12.2) ----

// A-fragment, 16x32 f16, from row-major A[m][k] with leading dim lda (halves).
__device__ __forceinline__ h16v16 load_a32(const h16* __restrict__ A, int lda) {
  const int l = threadIdx.x & 31;
  const int m = l & 15, sel = l >> 4;
  const h16* p = A + (size_t)m * lda + 8 * sel;
  h16v8 lo = *(const h16v8*)p;
  h16v8 hi = *(const h16v8*)(p + 16);
  h16v16 r;
#pragma unroll
  for (int i = 0; i < 8; ++i) { r[i] = lo[i]; r[i + 8] = hi[i]; }
  return r;
}

// B-fragment, 32x16 f16, from K-contiguous Bt[n][k] (i.e. B transposed), ld ldb.
__device__ __forceinline__ h16v16 load_b32(const h16* __restrict__ Bt, int ldb) {
  const int l = threadIdx.x & 31;
  const int n = l & 15, sel = l >> 4;
  const h16* p = Bt + (size_t)n * ldb + 8 * sel;
  h16v8 lo = *(const h16v8*)p;
  h16v8 hi = *(const h16v8*)(p + 16);
  h16v16 r;
#pragma unroll
  for (int i = 0; i < 8; ++i) { r[i] = lo[i]; r[i + 8] = hi[i]; }
  return r;
}

__device__ __forceinline__ f32v8 wmma32(h16v16 a, h16v16 b, f32v8 c) {
  return __builtin_amdgcn_wmma_f32_16x16x32_f16(false, a, false, b, (short)0, c,
                                                false, false);
}

__device__ __forceinline__ float sigf(float x) { return 1.f / (1.f + __expf(-x)); }

// ---------------- weight conversion -------------------------------------------------

__global__ void k_transpose_f16(const float* __restrict__ src, h16* __restrict__ dst,
                                int K, int N) {  // src[K][N] -> dst[N][K]
  int idx = blockIdx.x * blockDim.x + threadIdx.x;
  if (idx >= K * N) return;
  int k = idx / N, n = idx - k * N;
  dst[(size_t)n * K + k] = (h16)src[idx];
}

__global__ void k_convert_f16(const float* __restrict__ src, h16* __restrict__ dst, int n) {
  int i = blockIdx.x * blockDim.x + threadIdx.x;
  if (i < n) dst[i] = (h16)src[i];
}

// ---------------- prep: x = zn@l1, xT = time2vec, xe = x@enc ------------------------

__global__ void k_prep(const float* __restrict__ z,
                       const float* __restrict__ l1W, const float* __restrict__ l1b,
                       const float* __restrict__ w0, const float* __restrict__ b0,
                       const float* __restrict__ tw, const float* __restrict__ tb,
                       const float* __restrict__ eW, const float* __restrict__ eb,
                       float* __restrict__ x, float* __restrict__ xT,
                       float* __restrict__ xe) {
  __shared__ float xs[2];
  const int r = blockIdx.x;
  const int j = threadIdx.x;
  const float* zr = z + (size_t)r * (NOISE + TF);
  if (j < 2) {
    float a = l1b[j];
    for (int k = 0; k < NOISE; ++k) a += zr[k] * l1W[k * 2 + j];
    xs[j] = a;
    x[r * 2 + j] = a;
  }
  __syncthreads();
  const float* zt = zr + NOISE;
  float tv;
  if (j < EMB - 1) {
    float a = tb[j];
#pragma unroll
    for (int k = 0; k < TF; ++k) a += zt[k] * tw[k * (EMB - 1) + j];
    tv = sinf(a);
  } else {
    float a = b0[0];
#pragma unroll
    for (int k = 0; k < TF; ++k) a += zt[k] * w0[k];
    tv = a;
  }
  xT[(size_t)r * EMB + j] = tv;
  xe[(size_t)r * EMB + j] = eb[j] + xs[0] * eW[j] + xs[1] * eW[EMB + j];
}

// ---------------- layernorm -> f16 (one wave per row) -------------------------------

__global__ void k_layernorm_f16(const float* __restrict__ src, h16* __restrict__ dst,
                                const float* __restrict__ g, const float* __restrict__ b,
                                int nrows) {
  const int wave = threadIdx.x >> 5;
  const int row = blockIdx.x * (blockDim.x >> 5) + wave;
  if (row >= nrows) return;
  const float* xr = src + (size_t)row * EMB;
  const int l = threadIdx.x & 31;
  float v[10], s = 0.f, s2 = 0.f;
#pragma unroll
  for (int i = 0; i < 10; ++i) {
    float t = xr[l + i * 32];
    v[i] = t; s += t; s2 += t * t;
  }
#pragma unroll
  for (int off = 16; off > 0; off >>= 1) {
    s  += __shfl_xor(s,  off, 32);
    s2 += __shfl_xor(s2, off, 32);
  }
  const float mu = s * (1.f / EMB);
  const float rstd = rsqrtf(s2 * (1.f / EMB) - mu * mu + 1e-5f);
  h16* d = dst + (size_t)row * EMB;
#pragma unroll
  for (int i = 0; i < 10; ++i) {
    int j = l + i * 32;
    d[j] = (h16)((v[i] - mu) * rstd * g[j] + b[j]);
  }
}

// ---------------- LSTM scan (single workgroup, WMMA per step) -----------------------

__global__ void __launch_bounds__(256, 1)
k_lstm(const float* __restrict__ x, const float* __restrict__ Wih,
       const h16* __restrict__ Whh16, const float* __restrict__ h0,
       const float* __restrict__ c0, float* __restrict__ xh) {
  __shared__ __align__(16) h16 hbuf[16 * EMB];   // padded to M=16 (rows 8..15 zero)
  __shared__ float cbuf[8 * EMB];
  __shared__ float gbuf[8 * 4 * EMB];            // gates i|f|g|o
  __shared__ float xbuf[16];
  const int tid = threadIdx.x;
  const int wave = tid >> 5;
  const int l = tid & 31;
  const int g = l >> 4, nc = l & 15;

  for (int i = tid; i < 16 * EMB; i += 256) {
    int m = i / EMB, j = i - m * EMB;
    hbuf[i] = (m < 8) ? (h16)h0[m * EMB + j] : (h16)0.f;
  }
  for (int i = tid; i < 8 * EMB; i += 256) cbuf[i] = c0[i];
  __syncthreads();

  for (int s = 0; s < SEQ; ++s) {
    if (tid < 16) xbuf[tid] = x[((size_t)(tid >> 1) * SEQ + s) * 2 + (tid & 1)];
    __syncthreads();

    h16v16 af[10];
#pragma unroll
    for (int kc = 0; kc < 10; ++kc) af[kc] = load_a32(hbuf + kc * 32, EMB);

#pragma unroll 2
    for (int t = 0; t < 10; ++t) {            // wave owns N range [wave*160, +160)
      const int n0 = wave * 160 + t * 16;
      const h16* bp = Whh16 + (size_t)n0 * EMB;
      f32v8 acc = {};
#pragma unroll
      for (int kc = 0; kc < 10; ++kc)
        acc = wmma32(af[kc], load_b32(bp + kc * 32, EMB), acc);
      if (g == 0) {                           // only rows M=0..7 are real batches
#pragma unroll
        for (int r = 0; r < 8; ++r) {
          int n = n0 + nc;
          gbuf[r * 4 * EMB + n] =
              acc[r] + xbuf[r * 2] * Wih[n * 2] + xbuf[r * 2 + 1] * Wih[n * 2 + 1];
        }
      }
    }
    __syncthreads();

#pragma unroll
    for (int it = 0; it < 10; ++it) {         // 2560 (m,j) cells / 256 threads
      int idx = tid + it * 256;
      int m = idx / EMB, j = idx - m * EMB;
      float gi = gbuf[m * 4 * EMB + j];
      float gf = gbuf[m * 4 * EMB + EMB + j];
      float gg = gbuf[m * 4 * EMB + 2 * EMB + j];
      float go = gbuf[m * 4 * EMB + 3 * EMB + j];
      float cc = sigf(gf) * cbuf[idx] + sigf(gi) * tanhf(gg);
      float hh = sigf(go) * tanhf(cc);
      cbuf[idx] = cc;
      hbuf[m * EMB + j] = (h16)hh;
      xh[((size_t)m * SEQ + s) * EMB + j] = hh;
    }
    __syncthreads();
  }
}

// ---------------- generic WMMA GEMM with epilogues ----------------------------------
// C[ROWS x N] = A[ROWS x K](f16) * B, B given as Bt[N x K](f16) = W^T, + bias.
enum { EPI_Q = 0, EPI_VT = 1, EPI_PROJ = 2, EPI_GELU = 3, EPI_ADD = 4 };

__global__ void __launch_bounds__(256)
k_gemm(const h16* __restrict__ A, const h16* __restrict__ Bt,
       const float* __restrict__ bias, int K, int N,
       int epi, int part, h16* __restrict__ dh,
       float* __restrict__ df, const float* __restrict__ aux) {
  const int tid = threadIdx.x;
  const int wave = tid >> 5;
  const int mb = blockIdx.x * 128;
  const int nb = blockIdx.y * 64;
  const int mo = (wave >> 1) * 32, no = (wave & 1) * 32;
  const int l = tid & 31;
  const int g = l >> 4, nc = l & 15;

  f32v8 c00 = {}, c01 = {}, c10 = {}, c11 = {};
  const int nk = K >> 5;

#if HAVE_ASYNC
  // K-slice tiles double-buffered in LDS: A via per-lane async copies, B via TDM
  // (one descriptor per block per chunk) when available.
  __shared__ __align__(16) h16 Abuf[2][128 * 32];
  __shared__ __align__(16) h16 Bbuf[2][64 * 32];
  const int ar0 = tid >> 2, ao = (tid & 3) * 8;     // A: 512 16B chunks / 256 thr
  const int ar1 = ar0 + 64;

  auto stageA = [&](int kc, int buf) {
    const int ko = kc * 32;
    async_b128(A + (size_t)(mb + ar0) * K + ko + ao, &Abuf[buf][ar0 * 32 + ao]);
    async_b128(A + (size_t)(mb + ar1) * K + ko + ao, &Abuf[buf][ar1 * 32 + ao]);
  };
  auto stageB = [&](int kc, int buf) {
    const int ko = kc * 32;
#if HAVE_TDM
    if (wave == 0) tdm_load_2d(Bt + (size_t)nb * K + ko, &Bbuf[buf][0], 64, 32, K);
#else
    const int br = tid >> 2, bo = (tid & 3) * 8;    // B: 256 16B chunks / 256 thr
    async_b128(Bt + (size_t)(nb + br) * K + ko + bo, &Bbuf[buf][br * 32 + bo]);
#endif
  };

  stageA(0, 0);
  stageB(0, 0);
  for (int kc = 0; kc < nk; ++kc) {
    const int buf = kc & 1;
    if (kc + 1 < nk) {
      stageA(kc + 1, buf ^ 1);
      stageB(kc + 1, buf ^ 1);
#if HAVE_TDM
      __builtin_amdgcn_s_wait_tensorcnt(1);   // current B tile done (in-order TDM)
      __builtin_amdgcn_s_wait_asynccnt(2);    // current A chunks done
#else
      __builtin_amdgcn_s_wait_asynccnt(3);
#endif
    } else {
#if HAVE_TDM
      __builtin_amdgcn_s_wait_tensorcnt(0);
#endif
      __builtin_amdgcn_s_wait_asynccnt(0);
    }
    __syncthreads();                          // publish LDS tiles block-wide
    h16v16 a0 = load_a32(&Abuf[buf][(mo +  0) * 32], 32);
    h16v16 a1 = load_a32(&Abuf[buf][(mo + 16) * 32], 32);
    h16v16 b0 = load_b32(&Bbuf[buf][(no +  0) * 32], 32);
    h16v16 b1 = load_b32(&Bbuf[buf][(no + 16) * 32], 32);
    c00 = wmma32(a0, b0, c00);
    c01 = wmma32(a0, b1, c01);
    c10 = wmma32(a1, b0, c10);
    c11 = wmma32(a1, b1, c11);
    __syncthreads();                          // done reading before overwrite
  }
#else
  const h16* a0p = A + (size_t)(mb + mo) * K;
  const h16* a1p = A + (size_t)(mb + mo + 16) * K;
  const h16* b0p = Bt + (size_t)(nb + no) * K;
  const h16* b1p = Bt + (size_t)(nb + no + 16) * K;
  for (int kc = 0; kc < nk; ++kc) {
    const int ko = kc * 32;
    h16v16 a0 = load_a32(a0p + ko, K);
    h16v16 a1 = load_a32(a1p + ko, K);
    h16v16 b0 = load_b32(b0p + ko, K);
    h16v16 b1 = load_b32(b1p + ko, K);
    c00 = wmma32(a0, b0, c00);
    c01 = wmma32(a0, b1, c01);
    c10 = wmma32(a1, b0, c10);
    c11 = wmma32(a1, b1, c11);
    if (kc + 1 < nk) {
      __builtin_prefetch(a0p + ko + 32, 0, 0);
      __builtin_prefetch(b0p + ko + 32, 0, 0);
    }
  }
#endif

  auto epilog = [&](float acc, int M, int n) {
    float y = acc + bias[n];
    switch (epi) {
      case EPI_Q: {  // concat-head layout (B,H,S,192), slice `part`
        int b = M >> 10, s = M & 1023, h = n >> 6, dd = n & 63;
        dh[(((size_t)(b * NHEADS + h) * SEQ + s) * QKD) + part * HD + dd] = (h16)y;
      } break;
      case EPI_VT: { // V transposed: (B,H,64,S)
        int b = M >> 10, s = M & 1023, h = n >> 6, dd = n & 63;
        dh[((size_t)(b * NHEADS + h) * HD + dd) * SEQ + s] = (h16)y;
      } break;
      case EPI_PROJ:  // x_new = xe + proj(out)
        df[(size_t)M * EMB + n] = aux[(size_t)M * EMB + n] + y;
        break;
      case EPI_GELU:  // exact gelu -> f16
        dh[(size_t)M * FF + n] = (h16)(0.5f * y * (1.f + erff(y * 0.70710678118f)));
        break;
      case EPI_ADD:   // x_new += ff2(...)
        df[(size_t)M * EMB + n] += y;
        break;
    }
  };

#pragma unroll
  for (int r = 0; r < 8; ++r) {
    epilog(c00[r], mb + mo + r + 8 * g,      nb + no + nc);
    epilog(c01[r], mb + mo + r + 8 * g,      nb + no + 16 + nc);
    epilog(c10[r], mb + mo + 16 + r + 8 * g, nb + no + nc);
    epilog(c11[r], mb + mo + 16 + r + 8 * g, nb + no + 16 + nc);
  }
}

// ---------------- fused flash attention (one wave per 16-row Q tile) ----------------

__global__ void __launch_bounds__(256)
k_attn(const h16* __restrict__ Q, const h16* __restrict__ Kt,
       const h16* __restrict__ Vt, h16* __restrict__ out16) {
  __shared__ __align__(16) h16 pbuf[8][16 * 32];   // per-wave P relayout tile
  const int wave = threadIdx.x >> 5;
  const int task = blockIdx.x * 8 + wave;          // < 2560 exactly
  const int qt = task & 63;
  const int bh = task >> 6;
  const int b = bh / NHEADS, h = bh % NHEADS;
  const int l = threadIdx.x & 31;
  const int g = l >> 4, nc = l & 15;
  const float SC = 0.018633899812498247f;          // 1/(3*sqrt(320))

  const h16* Qp = Q + ((size_t)bh * SEQ + qt * 16) * QKD;
  h16v16 qf[6];
#pragma unroll
  for (int kc = 0; kc < 6; ++kc) qf[kc] = load_a32(Qp + kc * 32, QKD);

  f32v8 o0 = {}, o1 = {}, o2 = {}, o3 = {};
  float mst[8], lst[8];
#pragma unroll
  for (int r = 0; r < 8; ++r) { mst[r] = -1e30f; lst[r] = 0.f; }

  h16* pw = &pbuf[wave][0];
  const h16* Vbase = Vt + (size_t)bh * HD * SEQ;

  for (int kb = 0; kb < SEQ / 32; ++kb) {
    const int key0 = kb * 32;
    f32v8 s0 = {}, s1 = {};
    const h16* Kp0 = Kt + ((size_t)bh * SEQ + key0) * QKD;
    const h16* Kp1 = Kp0 + 16 * QKD;
#pragma unroll
    for (int kc = 0; kc < 6; ++kc) {
      s0 = wmma32(qf[kc], load_b32(Kp0 + kc * 32, QKD), s0);
      s1 = wmma32(qf[kc], load_b32(Kp1 + kc * 32, QKD), s1);
    }

    float p0[8], p1[8];
#pragma unroll
    for (int r = 0; r < 8; ++r) {
      const int qi = qt * 16 + r + 8 * g;
      float v0 = s0[r] * SC, v1 = s1[r] * SC;
      float d0 = (float)(key0 + nc - qi);
      float d1 = (float)(key0 + 16 + nc - qi);
      if (d0 > 0.f) v0 *= 1.f + __expf(-d0 * d0 * 0.2f);   // gamma band
      if (d1 > 0.f) v1 *= 1.f + __expf(-d1 * d1 * 0.2f);
      p0[r] = v0; p1[r] = v1;
    }

#pragma unroll
    for (int r = 0; r < 8; ++r) {            // online softmax per row (16-lane group)
      float m = fmaxf(p0[r], p1[r]);
      m = fmaxf(m, __shfl_xor(m, 1, 32));
      m = fmaxf(m, __shfl_xor(m, 2, 32));
      m = fmaxf(m, __shfl_xor(m, 4, 32));
      m = fmaxf(m, __shfl_xor(m, 8, 32));
      float mn = fmaxf(mst[r], m);
      float al = __expf(mst[r] - mn);
      mst[r] = mn;
      float e0 = __expf(p0[r] - mn);
      float e1 = __expf(p1[r] - mn);
      p0[r] = e0; p1[r] = e1;
      float ps = e0 + e1;
      ps += __shfl_xor(ps, 1, 32);
      ps += __shfl_xor(ps, 2, 32);
      ps += __shfl_xor(ps, 4, 32);
      ps += __shfl_xor(ps, 8, 32);
      lst[r] = lst[r] * al + ps;
      o0[r] *= al; o1[r] *= al; o2[r] *= al; o3[r] *= al;
    }

#pragma unroll
    for (int r = 0; r < 8; ++r) {            // C-frag -> A-frag via per-wave LDS
      int M = r + 8 * g;
      pw[M * 32 + nc]      = (h16)p0[r];
      pw[M * 32 + 16 + nc] = (h16)p1[r];
    }
    asm volatile("s_wait_dscnt 0" ::: "memory");  // DS in-order per wave
    h16v16 pa = load_a32(pw, 32);

    const h16* Vp = Vbase + key0;
    o0 = wmma32(pa, load_b32(Vp + (size_t)0  * SEQ, SEQ), o0);
    o1 = wmma32(pa, load_b32(Vp + (size_t)16 * SEQ, SEQ), o1);
    o2 = wmma32(pa, load_b32(Vp + (size_t)32 * SEQ, SEQ), o2);
    o3 = wmma32(pa, load_b32(Vp + (size_t)48 * SEQ, SEQ), o3);
  }

#pragma unroll
  for (int r = 0; r < 8; ++r) {
    float inv = 1.f / lst[r];
    int srow = qt * 16 + r + 8 * g;
    size_t base = ((size_t)b * SEQ + srow) * EMB + h * HD;
    out16[base + nc]      = (h16)(o0[r] * inv);
    out16[base + 16 + nc] = (h16)(o1[r] * inv);
    out16[base + 32 + nc] = (h16)(o2[r] * inv);
    out16[base + 48 + nc] = (h16)(o3[r] * inv);
  }
}

// ---------------- final head 320 -> 2 -----------------------------------------------

__global__ void k_head(const float* __restrict__ xnew, const float* __restrict__ W,
                       const float* __restrict__ b, float* __restrict__ out) {
  int t = blockIdx.x * blockDim.x + threadIdx.x;
  if (t >= ROWS * GEN) return;
  int r = t >> 1, gg = t & 1;
  const float* xp = xnew + (size_t)r * EMB;
  float acc = b[gg];
  for (int k = 0; k < EMB; ++k) acc += xp[k] * W[k * GEN + gg];
  out[t] = acc;
}

// ---------------- host orchestration -------------------------------------------------

extern "C" void kernel_launch(void* const* d_in, const int* in_sizes, int n_in,
                              void* d_out, int out_size, void* d_ws, size_t ws_size,
                              hipStream_t stream) {
  const float* z  = (const float*)d_in[0];
  const float* h0 = (const float*)d_in[1];
  const float* c0 = (const float*)d_in[2];

#define FP(i) ((const float*)d_in[(i)])
  const float *l1W, *l1b, *Wih, *Whh, *t2vw0, *t2vb0, *t2vw, *t2vb, *encW, *encb,
      *lng, *lnb, *ff1W, *ff1b, *ff2W, *ff2b, *l2W, *l2b, *qxW, *qxb, *kxW, *kxb,
      *vxW, *vxb, *qhW, *qhb, *khW, *khb, *qtW, *qtb, *ktW, *ktb, *projW, *projb;
  if (n_in >= 37 && in_sizes[3] == 256) {  // params flattened in insertion order
    l1W = FP(3);  l1b = FP(4);  Wih = FP(5);  Whh = FP(6);  t2vw0 = FP(7);
    t2vb0 = FP(8); t2vw = FP(9); t2vb = FP(10); encW = FP(11); encb = FP(12);
    lng = FP(13); lnb = FP(14); ff1W = FP(15); ff1b = FP(16); ff2W = FP(17);
    ff2b = FP(18); l2W = FP(19); l2b = FP(20); qxW = FP(21); qxb = FP(22);
    kxW = FP(23); kxb = FP(24); vxW = FP(25); vxb = FP(26); qhW = FP(27);
    qhb = FP(28); khW = FP(29); khb = FP(30); qtW = FP(31); qtb = FP(32);
    ktW = FP(33); ktb = FP(34); projW = FP(35); projb = FP(36);
  } else {                                  // params flattened alphabetically
    encW = FP(3);  encb = FP(4);  ff1W = FP(5);  ff1b = FP(6);  ff2W = FP(7);
    ff2b = FP(8);  khW = FP(9);  khb = FP(10); ktW = FP(11); ktb = FP(12);
    kxW = FP(13);  kxb = FP(14); l1W = FP(15); l1b = FP(16); l2W = FP(17);
    l2b = FP(18);  lnb = FP(19); lng = FP(20); Whh = FP(21); Wih = FP(22);
    projW = FP(23); projb = FP(24); qhW = FP(25); qhb = FP(26); qtW = FP(27);
    qtb = FP(28);  qxW = FP(29); qxb = FP(30); t2vb = FP(31); t2vb0 = FP(32);
    t2vw = FP(33); t2vw0 = FP(34); vxW = FP(35); vxb = FP(36);
  }
#undef FP

  char* wsb = (char*)d_ws;
  size_t off = 0;
  auto alloc = [&](size_t bytes) -> char* {
    char* p = wsb + off;
    off = (off + bytes + 255) & ~(size_t)255;
    return p;
  };
  float* xbuf  = (float*)alloc((size_t)ROWS * 2 * 4);
  float* xT    = (float*)alloc((size_t)ROWS * EMB * 4);
  float* xe    = (float*)alloc((size_t)ROWS * EMB * 4);
  float* xh    = (float*)alloc((size_t)ROWS * EMB * 4);
  float* xnew  = (float*)alloc((size_t)ROWS * EMB * 4);
  h16* xL16    = (h16*)alloc((size_t)ROWS * EMB * 2);
  h16* xH16    = (h16*)alloc((size_t)ROWS * EMB * 2);
  h16* xTn16   = (h16*)alloc((size_t)ROWS * EMB * 2);
  h16* Q16     = (h16*)alloc((size_t)RB * NHEADS * SEQ * QKD * 2);
  h16* K16     = (h16*)alloc((size_t)RB * NHEADS * SEQ * QKD * 2);
  h16* Vt16    = (h16*)alloc((size_t)RB * NHEADS * HD * SEQ * 2);
  h16* out16   = (h16*)alloc((size_t)ROWS * EMB * 2);
  h16* x1ln16  = (h16*)alloc((size_t)ROWS * EMB * 2);
  h16* ffmid16 = (h16*)alloc((size_t)ROWS * FF * 2);
  h16* wq   = (h16*)alloc((size_t)EMB * EMB * 2);
  h16* wk   = (h16*)alloc((size_t)EMB * EMB * 2);
  h16* wv   = (h16*)alloc((size_t)EMB * EMB * 2);
  h16* wqh  = (h16*)alloc((size_t)EMB * EMB * 2);
  h16* wkh  = (h16*)alloc((size_t)EMB * EMB * 2);
  h16* wqt  = (h16*)alloc((size_t)EMB * EMB * 2);
  h16* wkt  = (h16*)alloc((size_t)EMB * EMB * 2);
  h16* wpr  = (h16*)alloc((size_t)EMB * EMB * 2);
  h16* wf1t = (h16*)alloc((size_t)FF * EMB * 2);
  h16* wf2t = (h16*)alloc((size_t)EMB * FF * 2);
  h16* whh16 = (h16*)alloc((size_t)FF * EMB * 2);

  auto T = [&](const float* src, h16* dst, int K, int N) {
    int total = K * N;
    k_transpose_f16<<<(total + 255) / 256, 256, 0, stream>>>(src, dst, K, N);
  };
  T(qxW, wq, EMB, EMB);  T(kxW, wk, EMB, EMB);  T(vxW, wv, EMB, EMB);
  T(qhW, wqh, EMB, EMB); T(khW, wkh, EMB, EMB); T(qtW, wqt, EMB, EMB);
  T(ktW, wkt, EMB, EMB); T(projW, wpr, EMB, EMB);
  T(ff1W, wf1t, EMB, FF);
  T(ff2W, wf2t, FF, EMB);
  k_convert_f16<<<(FF * EMB + 255) / 256, 256, 0, stream>>>(Whh, whh16, FF * EMB);

  k_prep<<<ROWS, EMB, 0, stream>>>(z, l1W, l1b, t2vw0, t2vb0, t2vw, t2vb, encW, encb,
                                   xbuf, xT, xe);
  k_layernorm_f16<<<ROWS / 8, 256, 0, stream>>>(xe, xL16, lng, lnb, ROWS);
  k_layernorm_f16<<<ROWS / 8, 256, 0, stream>>>(xT, xH16, lng, lnb, ROWS);
  k_lstm<<<1, 256, 0, stream>>>(xbuf, Wih, whh16, h0, c0, xh);
  k_layernorm_f16<<<ROWS / 8, 256, 0, stream>>>(xh, xTn16, lng, lnb, ROWS);

  dim3 g320(ROWS / 128, EMB / 64);
  k_gemm<<<g320, 256, 0, stream>>>(xL16, wq, qxb, EMB, EMB, EPI_Q, 0, Q16, nullptr, nullptr);
  k_gemm<<<g320, 256, 0, stream>>>(xL16, wk, kxb, EMB, EMB, EPI_Q, 0, K16, nullptr, nullptr);
  k_gemm<<<g320, 256, 0, stream>>>(xL16, wv, vxb, EMB, EMB, EPI_VT, 0, Vt16, nullptr, nullptr);
  k_gemm<<<g320, 256, 0, stream>>>(xH16, wqh, qhb, EMB, EMB, EPI_Q, 1, Q16, nullptr, nullptr);
  k_gemm<<<g320, 256, 0, stream>>>(xH16, wkh, khb, EMB, EMB, EPI_Q, 1, K16, nullptr, nullptr);
  k_gemm<<<g320, 256, 0, stream>>>(xTn16, wqt, qtb, EMB, EMB, EPI_Q, 2, Q16, nullptr, nullptr);
  k_gemm<<<g320, 256, 0, stream>>>(xTn16, wkt, ktb, EMB, EMB, EPI_Q, 2, K16, nullptr, nullptr);

  k_attn<<<(RB * NHEADS * (SEQ / 16)) / 8, 256, 0, stream>>>(Q16, K16, Vt16, out16);

  k_gemm<<<g320, 256, 0, stream>>>(out16, wpr, projb, EMB, EMB, EPI_PROJ, 0, nullptr, xnew, xe);
  k_layernorm_f16<<<ROWS / 8, 256, 0, stream>>>(xnew, x1ln16, lng, lnb, ROWS);
  dim3 gff(ROWS / 128, FF / 64);
  k_gemm<<<gff, 256, 0, stream>>>(x1ln16, wf1t, ff1b, EMB, FF, EPI_GELU, 0, ffmid16, nullptr, nullptr);
  k_gemm<<<g320, 256, 0, stream>>>(ffmid16, wf2t, ff2b, FF, EMB, EPI_ADD, 0, nullptr, xnew, nullptr);

  k_head<<<(ROWS * GEN + 255) / 256, 256, 0, stream>>>(xnew, l2W, l2b, (float*)d_out);
}
